// SelfAttentionLayer_65481071405388
// MI455X (gfx1250) — matrix operand order repaired and checked by reading, not compiled, not run on hardware
//
#include <hip/hip_runtime.h>

// MI455X / gfx1250: wave32, WMMA 16x16x32 f16 -> f32 accumulate.
// Self-attention B=4, S=4096, D=1024. Compute-bound (~378 GFLOP vs ~15us of HBM
// traffic at 23.3 TB/s), so everything runs through v_wmma_f32_16x16x32_f16.

typedef _Float16 v16h __attribute__((ext_vector_type(16)));
typedef _Float16 v8h  __attribute__((ext_vector_type(8)));
typedef _Float16 v4h  __attribute__((ext_vector_type(4)));
typedef float    v8f  __attribute__((ext_vector_type(8)));

#define BB 4
#define SS 4096
#define DD 1024

static __device__ inline v8f wmma16(v16h a, v16h b, v8f c) {
  // (neg_a, A, neg_b, B, c_mod, C, reuse_a, reuse_b)
  return __builtin_amdgcn_wmma_f32_16x16x32_f16(false, a, false, b, (short)0, c,
                                                false, false);
}

// A-fragment 16x32 (MxK), f16, from row-major memory with row stride ld.
// Contiguous k order (not ISA table order): lanes 0-15 hold M=lane, K=0..15; lanes 16-31 hold
// M=lane-16, K=16..31. Two contiguous 16B chunks per lane.
static __device__ inline v16h load_a_frag(const _Float16* p, int ld, int lane) {
  const int m    = lane & 15;
  const int koff = (lane >> 4) << 4;  // 0 or 16
  const _Float16* r = p + (size_t)m * ld + koff;
  v8h lo = *(const v8h*)(r);
  v8h hi = *(const v8h*)(r + 8);
  v16h a;
#pragma unroll
  for (int i = 0; i < 8; ++i) { a[i] = lo[i]; a[i + 8] = hi[i]; }
  return a;
}

// B-fragment 32x16 (KxN), f16, from memory stored as [n][k] (each column of B
// is a contiguous run of K). Lanes 0-15: N=lane, K=0..15; lanes 16-31:
// N=lane-16, K=16..31. One contiguous 32B run per lane.
static __device__ inline v16h load_b_frag(const _Float16* p, int ld, int lane) {
  const int n    = lane & 15;
  const int koff = (lane >> 4) << 4;  // 0 or 16
  return *(const v16h*)(p + (size_t)n * ld + koff);
}

// ---------------------------------------------------------------------------
// Kernel 0: fp32 -> f16 conversion (vectorized x4)
// ---------------------------------------------------------------------------
__global__ __launch_bounds__(256) void cvt_f32_f16(const float* __restrict__ in,
                                                   _Float16* __restrict__ out,
                                                   int n4) {
  int i = blockIdx.x * 256 + threadIdx.x;
  if (i < n4) {
    float4 v = ((const float4*)in)[i];
    v4h o;
    o[0] = (_Float16)v.x; o[1] = (_Float16)v.y;
    o[2] = (_Float16)v.z; o[3] = (_Float16)v.w;
    ((v4h*)out)[i] = o;
  }
}

// ---------------------------------------------------------------------------
// Kernel 1: QKV projection GEMM, C = x @ W^T  (x:(B*S,D) f16, W:(D,D) f16).
// blockIdx.z selects Wq/Wk/Wv. z==2 stores V transposed as Vt (B, D, S) so the
// attention kernel's P@V B-fragments are contiguous loads.
// Block = 256 thr (8 waves), tile 64(M) x 128(N); each wave does a 32x32
// subtile = 2x2 WMMA accumulators, K-loop of 32 over D=1024.
// ---------------------------------------------------------------------------
__global__ __launch_bounds__(256) void qkv_gemm(
    const _Float16* __restrict__ xh, const _Float16* __restrict__ Wh,
    _Float16* __restrict__ Qh, _Float16* __restrict__ Kh,
    _Float16* __restrict__ Vt) {
  const int lane = threadIdx.x & 31;
  const int w    = threadIdx.x >> 5;
  const int z    = blockIdx.z;
  const _Float16* W = Wh + (size_t)z * DD * DD;

  const int m0 = blockIdx.x * 64 + (w & 1) * 32;
  const int n0 = blockIdx.y * 128 + (w >> 1) * 32;

  v8f acc[2][2];
#pragma unroll
  for (int i = 0; i < 2; ++i)
#pragma unroll
    for (int j = 0; j < 2; ++j)
#pragma unroll
      for (int g = 0; g < 8; ++g) acc[i][j][g] = 0.f;

  for (int k0 = 0; k0 < DD; k0 += 32) {
    v16h a0 = load_a_frag(xh + (size_t)m0 * DD + k0, DD, lane);
    v16h a1 = load_a_frag(xh + (size_t)(m0 + 16) * DD + k0, DD, lane);
    v16h b0 = load_b_frag(W + (size_t)n0 * DD + k0, DD, lane);
    v16h b1 = load_b_frag(W + (size_t)(n0 + 16) * DD + k0, DD, lane);
    acc[0][0] = wmma16(a0, b0, acc[0][0]);
    acc[0][1] = wmma16(a0, b1, acc[0][1]);
    acc[1][0] = wmma16(a1, b0, acc[1][0]);
    acc[1][1] = wmma16(a1, b1, acc[1][1]);
  }

  // C/D layout: VGPR g -> row g + (lane/16)*8, col lane%16.
  const int hi = lane >> 4, nl = lane & 15;
#pragma unroll
  for (int i = 0; i < 2; ++i)
#pragma unroll
    for (int j = 0; j < 2; ++j)
#pragma unroll
      for (int g = 0; g < 8; ++g) {
        const int m = m0 + 16 * i + g + hi * 8;
        const int n = n0 + 16 * j + nl;
        const _Float16 v = (_Float16)acc[i][j][g];
        if (z == 0) {
          Qh[(size_t)m * DD + n] = v;
        } else if (z == 1) {
          Kh[(size_t)m * DD + n] = v;
        } else {
          const int b = m >> 12, s = m & (SS - 1);
          Vt[(size_t)b * DD * SS + (size_t)n * SS + s] = v;
        }
      }
}

// ---------------------------------------------------------------------------
// Kernel 2: flash attention. Block = 16 queries x full D. 8 waves; wave w owns
// d-slice [w*128, w*128+128) -> 8 f32 C-fragments (64 VGPRs). Per 128-key tile:
//   wave w computes the 16x16 score tile for keys [t0+16w, t0+16w+16)
//   (32 WMMAs over D), online softmax via LDS + lane shuffles, then 32 WMMAs
//   of P @ V into the running accumulators. K/V stream from L2 (fully
//   resident: 64 MB of K+V vs 192 MB L2).
// ---------------------------------------------------------------------------
__global__ __launch_bounds__(256) void flash_attn(
    const _Float16* __restrict__ Qh, const _Float16* __restrict__ Kh,
    const _Float16* __restrict__ Vt, float* __restrict__ out) {
  __shared__ alignas(32) _Float16 Qs[16 * DD];    // 32 KB
  __shared__ alignas(32) float    Ssc[16 * 128];  // 8 KB raw scores
  __shared__ alignas(32) _Float16 Ps[16 * 128];   // 4 KB exp(scores)
  __shared__ float mrow[16], lrow[16], arow[16];

  const int lane  = threadIdx.x & 31;
  const int w     = threadIdx.x >> 5;
  const int b     = blockIdx.y;
  const int qbase = blockIdx.x * 16;
  const int d0    = w * 128;

  // Stage the 16x1024 Q tile (contiguous in memory) into LDS.
  const _Float16* qsrc = Qh + ((size_t)b * SS + qbase) * DD;
  for (int i = threadIdx.x; i < 16 * DD / 8; i += 256)
    *(v8h*)(&Qs[i * 8]) = *(const v8h*)(qsrc + i * 8);
  if (threadIdx.x < 16) { mrow[threadIdx.x] = -1e30f; lrow[threadIdx.x] = 0.f; }
  __syncthreads();

  v8f acc[8];
#pragma unroll
  for (int n = 0; n < 8; ++n)
#pragma unroll
    for (int g = 0; g < 8; ++g) acc[n][g] = 0.f;

  const float scale = 0.03125f;  // 1/sqrt(1024)
  const int hi = lane >> 4, nl = lane & 15;

  for (int t0 = 0; t0 < SS; t0 += 128) {
    // ---- scores: wave w -> keys [t0+16w, t0+16w+16) ----
    v8f sc;
#pragma unroll
    for (int g = 0; g < 8; ++g) sc[g] = 0.f;
    const _Float16* kbase = Kh + ((size_t)b * SS + t0 + w * 16) * DD;
    for (int k0 = 0; k0 < DD; k0 += 32) {
      v16h a  = load_a_frag(Qs + k0, DD, lane);        // from LDS
      v16h bk = load_b_frag(kbase + k0, DD, lane);     // from L2
      sc = wmma16(a, bk, sc);
    }
#pragma unroll
    for (int g = 0; g < 8; ++g)
      Ssc[(g + hi * 8) * 128 + w * 16 + nl] = sc[g] * scale;
    __syncthreads();

    // ---- online softmax: wave w handles rows 2w, 2w+1; 16 lanes/row ----
    {
      const int r = 2 * w + hi;
      const int j = nl;
      float sv[8];
      float lm = -1e30f;
#pragma unroll
      for (int t = 0; t < 8; ++t) {
        sv[t] = Ssc[r * 128 + j + 16 * t];
        lm = fmaxf(lm, sv[t]);
      }
#pragma unroll
      for (int o = 1; o < 16; o <<= 1) lm = fmaxf(lm, __shfl_xor(lm, o, 16));
      const float mo = mrow[r];
      const float nm = fmaxf(mo, lm);
      float ps = 0.f;
#pragma unroll
      for (int t = 0; t < 8; ++t) {
        float p = __expf(sv[t] - nm);
        Ps[r * 128 + j + 16 * t] = (_Float16)p;
        ps += p;
      }
#pragma unroll
      for (int o = 1; o < 16; o <<= 1) ps += __shfl_xor(ps, o, 16);
      if (j == 0) {
        const float al = __expf(mo - nm);
        lrow[r] = lrow[r] * al + ps;
        mrow[r] = nm;
        arow[r] = al;
      }
    }
    __syncthreads();

    // ---- rescale accumulators by alpha(row) ----
    {
      float al[8];
#pragma unroll
      for (int g = 0; g < 8; ++g) al[g] = arow[g + hi * 8];
#pragma unroll
      for (int n = 0; n < 8; ++n)
#pragma unroll
        for (int g = 0; g < 8; ++g) acc[n][g] *= al[g];
    }

    // ---- O += P(16x128) @ V(128 x d-slice); Vt is (B,D,S) so B-frags are
    //      contiguous 32B runs per lane ----
    const _Float16* vb = Vt + (size_t)b * DD * SS + (size_t)d0 * SS + t0;
#pragma unroll
    for (int kk = 0; kk < 128; kk += 32) {
      v16h a = load_a_frag(Ps + kk, 128, lane);  // from LDS
#pragma unroll
      for (int n = 0; n < 8; ++n) {
        v16h bv = load_b_frag(vb + (size_t)(n * 16) * SS + kk, SS, lane);
        acc[n] = wmma16(a, bv, acc[n]);
      }
    }
    __syncthreads();
  }

  // ---- normalize and store fp32 output ----
  {
    float inv[8];
#pragma unroll
    for (int g = 0; g < 8; ++g) inv[g] = 1.f / lrow[g + hi * 8];
#pragma unroll
    for (int n = 0; n < 8; ++n)
#pragma unroll
      for (int g = 0; g < 8; ++g) {
        const int s = qbase + g + hi * 8;
        const int c = d0 + n * 16 + nl;
        out[((size_t)b * SS + s) * DD + c] = acc[n][g] * inv[g];
      }
  }
}

// ---------------------------------------------------------------------------
extern "C" void kernel_launch(void* const* d_in, const int* in_sizes, int n_in,
                              void* d_out, int out_size, void* d_ws,
                              size_t ws_size, hipStream_t stream) {
  (void)in_sizes; (void)n_in; (void)out_size; (void)ws_size;
  const float* x  = (const float*)d_in[0];
  const float* Wq = (const float*)d_in[1];
  const float* Wk = (const float*)d_in[2];
  const float* Wv = (const float*)d_in[3];
  float* out = (float*)d_out;

  // Workspace carve-up (~134 MB, all f16):
  char* ws = (char*)d_ws;
  const size_t nX = (size_t)BB * SS * DD;      // 16.8M elems
  _Float16* xh = (_Float16*)ws; ws += nX * 2;
  _Float16* Wh = (_Float16*)ws; ws += (size_t)3 * DD * DD * 2;
  _Float16* Qh = (_Float16*)ws; ws += nX * 2;
  _Float16* Kh = (_Float16*)ws; ws += nX * 2;
  _Float16* Vt = (_Float16*)ws;

  // 0) fp32 -> f16
  cvt_f32_f16<<<(int)(nX / 4 / 256), 256, 0, stream>>>(x, xh, (int)(nX / 4));
  cvt_f32_f16<<<DD * DD / 4 / 256, 256, 0, stream>>>(Wq, Wh, DD * DD / 4);
  cvt_f32_f16<<<DD * DD / 4 / 256, 256, 0, stream>>>(Wk, Wh + (size_t)DD * DD,
                                                     DD * DD / 4);
  cvt_f32_f16<<<DD * DD / 4 / 256, 256, 0, stream>>>(Wv, Wh + (size_t)2 * DD * DD,
                                                     DD * DD / 4);

  // 1) Q, K, V projections (V stored transposed)
  qkv_gemm<<<dim3((BB * SS) / 64, DD / 128, 3), 256, 0, stream>>>(xh, Wh, Qh,
                                                                  Kh, Vt);

  // 2) flash attention
  flash_attn<<<dim3(SS / 16, BB), 256, 0, stream>>>(Qh, Kh, Vt, out);
}
